// MultiheadSelfAttention_36687610642557
// MI455X (gfx1250) — compile-verified, matrix operand order checked by reading, not run on hardware
//
#include <hip/hip_runtime.h>
#include <hip/hip_bf16.h>
#include <stdint.h>

#define D_MODEL 1024
#define N_HEAD  16
#define D_HEAD  64
#define SEQ     2048
#define BATCH   4
#define M_TOT   (BATCH * SEQ)   // 8192

typedef __attribute__((ext_vector_type(16))) __bf16 v16bf;
typedef __attribute__((ext_vector_type(8)))  float  v8f;

union Frag16 {
    v16bf v;
    uint4 q[2];
    unsigned short h[16];
};

__device__ __forceinline__ unsigned short f2bf(float f) {
    union { float f; unsigned int u; } c; c.f = f;
    unsigned int u = c.u;
    unsigned int r = (u + 0x7FFFu + ((u >> 16) & 1u)) >> 16;  // round-to-nearest-even
    return (unsigned short)r;
}

// Async global -> LDS copy of 16 bytes (tracked by ASYNCcnt).
__device__ __forceinline__ void async_copy_b128(const void* gaddr, const void* lds) {
    unsigned lds_off = (unsigned)(uintptr_t)lds;       // low 32 bits = LDS offset
    unsigned long long ga = (unsigned long long)(uintptr_t)gaddr;
    asm volatile("global_load_async_to_lds_b128 %0, %1, off"
                 :: "v"(lds_off), "v"(ga) : "memory");
}
__device__ __forceinline__ void wait_async0() {
    asm volatile("s_wait_asynccnt 0x0" ::: "memory");
}

// ---------------------------------------------------------------------------
// fp32 -> bf16 conversion (grid-stride)
// ---------------------------------------------------------------------------
__global__ void cvt_f32_bf16(const float* __restrict__ src,
                             unsigned short* __restrict__ dst, int n) {
    int i = blockIdx.x * blockDim.x + threadIdx.x;
    int stride = gridDim.x * blockDim.x;
    for (; i < n; i += stride) dst[i] = f2bf(src[i]);
}

// ---------------------------------------------------------------------------
// GEMM: Y[m,n] = sum_k A[m,k] * W[n,k]   (A: 8192x1024 bf16, W: 1024x1024 bf16)
// mode 0: bf16 head-split  out[((b*16+h)*SEQ+s)*64+d]
// mode 1: bf16 head-split transposed out[((b*16+h)*64+d)*SEQ+s]   (for V)
// mode 2: fp32 row-major   out[m*1024+n]
// Block: 256 thr = 8 waves as 4(M) x 2(N); block tile 128x64; wave tile 32x32.
// Double-buffered LDS (BK=64 -> 8 WMMAs per sync section), filled by
// global_load_async_to_lds_b128 (ASYNCcnt path).
// ---------------------------------------------------------------------------
#define BM 128
#define BN 64
#define BK 64
#define LDT 72   // BK + 8 bf16 pad

__global__ __launch_bounds__(256) void gemm_xwt(
    const unsigned short* __restrict__ A,
    const unsigned short* __restrict__ W,
    void* __restrict__ out, int mode)
{
    __shared__ unsigned short As[2][BM * LDT];
    __shared__ unsigned short Bs[2][BN * LDT];

    const int m0   = blockIdx.y * BM;
    const int n0   = blockIdx.x * BN;
    const int tid  = threadIdx.x;
    const int wave = tid >> 5;
    const int lane = tid & 31;
    const int wm   = wave & 3;       // 0..3 (M)
    const int wn   = wave >> 2;      // 0..1 (N)
    const int l16  = lane & 15;
    const int half = lane >> 4;

    v8f acc[2][2];
#pragma unroll
    for (int i = 0; i < 2; i++)
#pragma unroll
        for (int j = 0; j < 2; j++)
#pragma unroll
            for (int r = 0; r < 8; r++) acc[i][j][r] = 0.0f;

    // staging: A tile 128x64 bf16 = 1024 x 16B chunks (4/thread);
    //          B tile  64x64 bf16 =  512 chunks (2/thread)
    auto stage = [&](int buf, int k0) {
#pragma unroll
        for (int i = 0; i < 4; i++) {
            int c = tid + i * 256;
            int row = c >> 3, cc = c & 7;
            async_copy_b128(A + (size_t)(m0 + row) * D_MODEL + k0 + cc * 8,
                            &As[buf][row * LDT + cc * 8]);
        }
#pragma unroll
        for (int i = 0; i < 2; i++) {
            int c = tid + i * 256;
            int row = c >> 3, cc = c & 7;
            async_copy_b128(W + (size_t)(n0 + row) * D_MODEL + k0 + cc * 8,
                            &Bs[buf][row * LDT + cc * 8]);
        }
    };

    stage(0, 0);
    wait_async0();
    __syncthreads();

    for (int k0 = 0; k0 < D_MODEL; k0 += BK) {
        const int buf = (k0 >> 6) & 1;
        if (k0 + BK < D_MODEL) stage(buf ^ 1, k0 + BK);   // overlap with compute

        Frag16 af[2][2], bfr[2][2];   // [tile][kstep]
#pragma unroll
        for (int mi = 0; mi < 2; mi++)
#pragma unroll
            for (int ks = 0; ks < 2; ks++) {
                const unsigned short* p =
                    &As[buf][(wm * 32 + mi * 16 + l16) * LDT + ks * 32 + half * 8];
                af[mi][ks].q[0] = *(const uint4*)(p);
                af[mi][ks].q[1] = *(const uint4*)(p + 16);
            }
#pragma unroll
        for (int ni = 0; ni < 2; ni++)
#pragma unroll
            for (int ks = 0; ks < 2; ks++) {
                const unsigned short* p =
                    &Bs[buf][(wn * 32 + ni * 16 + l16) * LDT + ks * 32 + half * 8];
                bfr[ni][ks].q[0] = *(const uint4*)(p);
                bfr[ni][ks].q[1] = *(const uint4*)(p + 16);
            }
#pragma unroll
        for (int ks = 0; ks < 2; ks++)
#pragma unroll
            for (int mi = 0; mi < 2; mi++)
#pragma unroll
                for (int ni = 0; ni < 2; ni++)
                    acc[mi][ni] = __builtin_amdgcn_wmma_f32_16x16x32_bf16(
                        false, af[mi][ks].v, false, bfr[ni][ks].v, (short)0,
                        acc[mi][ni], false, false);

        wait_async0();      // next buffer filled
        __syncthreads();    // all waves done reading current buffer
    }

#pragma unroll
    for (int mi = 0; mi < 2; mi++)
#pragma unroll
        for (int ni = 0; ni < 2; ni++)
#pragma unroll
            for (int r = 0; r < 8; r++) {
                int m = m0 + wm * 32 + mi * 16 + half * 8 + r;
                int n = n0 + wn * 32 + ni * 16 + l16;
                float v = acc[mi][ni][r];
                if (mode == 2) {
                    ((float*)out)[(size_t)m * D_MODEL + n] = v;
                } else {
                    int b = m >> 11, s = m & (SEQ - 1);
                    int h = n >> 6,  d = n & (D_HEAD - 1);
                    unsigned short* o = (unsigned short*)out;
                    if (mode == 0)
                        o[(((size_t)(b * N_HEAD + h)) * SEQ + s) * D_HEAD + d] = f2bf(v);
                    else
                        o[(((size_t)(b * N_HEAD + h)) * D_HEAD + d) * SEQ + s] = f2bf(v);
                }
            }
}

// ---------------------------------------------------------------------------
// Causal flash attention. Q,K: [64][2048][64] bf16;  Vt: [64][64][2048] bf16.
// Block = 128 thr = 4 waves; wave owns 32 query rows; block tile = 128 rows.
// Ctx out: [4][2048][1024] bf16 (head-merged, ready for W_O GEMM).
// ---------------------------------------------------------------------------
#define AW 4

__global__ __launch_bounds__(128) void attn_causal(
    const unsigned short* __restrict__ Q,
    const unsigned short* __restrict__ K,
    const unsigned short* __restrict__ Vt,
    unsigned short* __restrict__ Ctx)
{
    __shared__ float Pld[AW][32][33];

    const int bh   = blockIdx.y;
    const int b    = bh >> 4, h = bh & 15;
    const int i0   = blockIdx.x * (AW * 32);
    const int wave = threadIdx.x >> 5;
    const int lane = threadIdx.x & 31;
    const int l16  = lane & 15, half = lane >> 4;
    const int rb   = i0 + wave * 32;

    const unsigned short* Qb = Q  + (size_t)bh * SEQ * D_HEAD;
    const unsigned short* Kb = K  + (size_t)bh * SEQ * D_HEAD;
    const unsigned short* Vb = Vt + (size_t)bh * D_HEAD * SEQ;

    // Q fragments for this wave's 32 rows: [m tile][k step]
    Frag16 qf[2][2];
#pragma unroll
    for (int mi = 0; mi < 2; mi++)
#pragma unroll
        for (int ks = 0; ks < 2; ks++) {
            const unsigned short* p =
                Qb + (size_t)(rb + mi * 16 + l16) * D_HEAD + ks * 32 + half * 8;
            qf[mi][ks].q[0] = *(const uint4*)(p);
            qf[mi][ks].q[1] = *(const uint4*)(p + 16);
        }

    v8f Oacc[2][4];
    float mrun[2][8], lrun[2][8];
#pragma unroll
    for (int mi = 0; mi < 2; mi++) {
#pragma unroll
        for (int nt = 0; nt < 4; nt++)
#pragma unroll
            for (int r = 0; r < 8; r++) Oacc[mi][nt][r] = 0.0f;
#pragma unroll
        for (int r = 0; r < 8; r++) { mrun[mi][r] = -3.0e38f; lrun[mi][r] = 0.0f; }
    }

    const int njt = (i0 >> 5) + AW;    // uniform over block (extra tiles fully masked)
    for (int jt = 0; jt < njt; jt++) {
        const int jb = jt * 32;

        // --- S = Q K^T -------------------------------------------------------
        Frag16 kf[2][2];
#pragma unroll
        for (int ni = 0; ni < 2; ni++)
#pragma unroll
            for (int ks = 0; ks < 2; ks++) {
                const unsigned short* p =
                    Kb + (size_t)(jb + ni * 16 + l16) * D_HEAD + ks * 32 + half * 8;
                kf[ni][ks].q[0] = *(const uint4*)(p);
                kf[ni][ks].q[1] = *(const uint4*)(p + 16);
            }
        v8f S[2][2];
#pragma unroll
        for (int mi = 0; mi < 2; mi++)
#pragma unroll
            for (int ni = 0; ni < 2; ni++) {
#pragma unroll
                for (int r = 0; r < 8; r++) S[mi][ni][r] = 0.0f;
#pragma unroll
                for (int ks = 0; ks < 2; ks++)
                    S[mi][ni] = __builtin_amdgcn_wmma_f32_16x16x32_bf16(
                        false, qf[mi][ks].v, false, kf[ni][ks].v, (short)0,
                        S[mi][ni], false, false);
            }

        // --- scale + causal mask --------------------------------------------
#pragma unroll
        for (int mi = 0; mi < 2; mi++)
#pragma unroll
            for (int ni = 0; ni < 2; ni++)
#pragma unroll
                for (int r = 0; r < 8; r++) {
                    int mg = rb + mi * 16 + half * 8 + r;
                    int ng = jb + ni * 16 + l16;
                    float v = S[mi][ni][r] * 0.125f;
                    S[mi][ni][r] = (ng > mg) ? -3.0e38f : v;
                }

        // --- online softmax --------------------------------------------------
        float mnew[2][8];
#pragma unroll
        for (int mi = 0; mi < 2; mi++)
#pragma unroll
            for (int r = 0; r < 8; r++) {
                float x = fmaxf(S[mi][0][r], S[mi][1][r]);
#pragma unroll
                for (int off = 1; off < 16; off <<= 1)
                    x = fmaxf(x, __shfl_xor(x, off, 32));
                mnew[mi][r] = fmaxf(mrun[mi][r], x);
            }
#pragma unroll
        for (int mi = 0; mi < 2; mi++)
#pragma unroll
            for (int r = 0; r < 8; r++) {
                float a = __expf(mrun[mi][r] - mnew[mi][r]);
                mrun[mi][r] = mnew[mi][r];
                lrun[mi][r] *= a;
#pragma unroll
                for (int nt = 0; nt < 4; nt++) Oacc[mi][nt][r] *= a;
            }
#pragma unroll
        for (int mi = 0; mi < 2; mi++)
#pragma unroll
            for (int ni = 0; ni < 2; ni++)
#pragma unroll
                for (int r = 0; r < 8; r++)
                    S[mi][ni][r] = __expf(S[mi][ni][r] - mnew[mi][r]);
#pragma unroll
        for (int mi = 0; mi < 2; mi++)
#pragma unroll
            for (int r = 0; r < 8; r++) {
                float s = S[mi][0][r] + S[mi][1][r];
#pragma unroll
                for (int off = 1; off < 16; off <<= 1)
                    s += __shfl_xor(s, off, 32);
                lrun[mi][r] += s;
            }

        // --- transpose P (C layout -> A layout) through LDS ------------------
#pragma unroll
        for (int mi = 0; mi < 2; mi++)
#pragma unroll
            for (int ni = 0; ni < 2; ni++)
#pragma unroll
                for (int r = 0; r < 8; r++)
                    Pld[wave][mi * 16 + half * 8 + r][ni * 16 + l16] = S[mi][ni][r];
        __syncthreads();

        Frag16 pf[2];
#pragma unroll
        for (int mi = 0; mi < 2; mi++) {
#pragma unroll
            for (int t = 0; t < 8; t++) {
                pf[mi].h[t]     = f2bf(Pld[wave][mi * 16 + l16][half * 8 + t]);
                pf[mi].h[8 + t] = f2bf(Pld[wave][mi * 16 + l16][16 + half * 8 + t]);
            }
        }
        __syncthreads();

        // --- O += P V  (V^T stored so B fragments load contiguously) ---------
#pragma unroll
        for (int nt = 0; nt < 4; nt++) {
            Frag16 vf;
            const unsigned short* p =
                Vb + (size_t)(nt * 16 + l16) * SEQ + jb + half * 8;
            vf.q[0] = *(const uint4*)(p);
            vf.q[1] = *(const uint4*)(p + 16);
#pragma unroll
            for (int mi = 0; mi < 2; mi++)
                Oacc[mi][nt] = __builtin_amdgcn_wmma_f32_16x16x32_bf16(
                    false, pf[mi].v, false, vf.v, (short)0, Oacc[mi][nt], false, false);
        }
    }

    // --- epilogue: normalize, store head-merged bf16 context -----------------
#pragma unroll
    for (int mi = 0; mi < 2; mi++)
#pragma unroll
        for (int nt = 0; nt < 4; nt++)
#pragma unroll
            for (int r = 0; r < 8; r++) {
                int s = rb + mi * 16 + half * 8 + r;
                int d = nt * 16 + l16;
                float v = Oacc[mi][nt][r] / lrun[mi][r];
                Ctx[((size_t)(b * SEQ + s)) * D_MODEL + h * D_HEAD + d] = f2bf(v);
            }
}

// ---------------------------------------------------------------------------
extern "C" void kernel_launch(void* const* d_in, const int* in_sizes, int n_in,
                              void* d_out, int out_size, void* d_ws, size_t ws_size,
                              hipStream_t stream) {
    const float* x  = (const float*)d_in[0];
    const float* wq = (const float*)d_in[1];
    const float* wk = (const float*)d_in[2];
    const float* wv = (const float*)d_in[3];
    const float* wo = (const float*)d_in[4];

    char* p = (char*)d_ws;
    unsigned short* xb  = (unsigned short*)p; p += (size_t)M_TOT * D_MODEL * 2;
    unsigned short* wqb = (unsigned short*)p; p += (size_t)D_MODEL * D_MODEL * 2;
    unsigned short* wkb = (unsigned short*)p; p += (size_t)D_MODEL * D_MODEL * 2;
    unsigned short* wvb = (unsigned short*)p; p += (size_t)D_MODEL * D_MODEL * 2;
    unsigned short* wob = (unsigned short*)p; p += (size_t)D_MODEL * D_MODEL * 2;
    unsigned short* q   = (unsigned short*)p; p += (size_t)M_TOT * D_MODEL * 2;
    unsigned short* k   = (unsigned short*)p; p += (size_t)M_TOT * D_MODEL * 2;
    unsigned short* vt  = (unsigned short*)p; p += (size_t)M_TOT * D_MODEL * 2;
    unsigned short* ctx = (unsigned short*)p; p += (size_t)M_TOT * D_MODEL * 2;

    cvt_f32_bf16<<<2048, 256, 0, stream>>>(x, xb, M_TOT * D_MODEL);
    cvt_f32_bf16<<<512, 256, 0, stream>>>(wq, wqb, D_MODEL * D_MODEL);
    cvt_f32_bf16<<<512, 256, 0, stream>>>(wk, wkb, D_MODEL * D_MODEL);
    cvt_f32_bf16<<<512, 256, 0, stream>>>(wv, wvb, D_MODEL * D_MODEL);
    cvt_f32_bf16<<<512, 256, 0, stream>>>(wo, wob, D_MODEL * D_MODEL);

    dim3 gg(D_MODEL / BN, M_TOT / BM);
    gemm_xwt<<<gg, 256, 0, stream>>>(xb, wqb, q,  0);
    gemm_xwt<<<gg, 256, 0, stream>>>(xb, wkb, k,  0);
    gemm_xwt<<<gg, 256, 0, stream>>>(xb, wvb, vt, 1);

    dim3 ga(SEQ / (AW * 32), BATCH * N_HEAD);
    attn_causal<<<ga, 128, 0, stream>>>(q, k, vt, ctx);

    gemm_xwt<<<gg, 256, 0, stream>>>(ctx, wob, d_out, 2);
}
